// GatedLinearAttentionBlock_19825569038707
// MI455X (gfx1250) — matrix operand order repaired and checked by reading, not compiled, not run on hardware
//
#include <hip/hip_runtime.h>

typedef __attribute__((ext_vector_type(16))) __bf16 v16bf;
typedef __attribute__((ext_vector_type(8)))  float  v8f;

#define WMMA_BF16(a, b, c) \
  __builtin_amdgcn_wmma_f32_16x16x32_bf16(false, (a), false, (b), (short)0, (c), false, false)

// ---- inverse fragment mappings (CDNA5 wave32, 16x16x32 bf16) ----
// A tile 16x32: element (m,k) -> lane m + 16*(k>>4), elem k&15
static __device__ __forceinline__ int a_lane(int m, int k) { return m + 16 * (k >> 4); }
static __device__ __forceinline__ int a_elem(int k) {
  return k & 15;
}
// B tile 32x16: element (k,n) -> lane n + 16*(k>>4), elem k&15
static __device__ __forceinline__ int b_lane(int n, int k) { return n + 16 * (k >> 4); }

static __device__ __forceinline__ v16bf frag_load(const __bf16* p) {
  return *(const v16bf*)p;
}

// ---------------- Kernel 0: weights f32 -> bf16, B-fragment-swizzled ----------------
// Wsw layout: tile t = which*32 + kk*8 + nt ; per tile 32 lanes x 16 contiguous bf16
__global__ void wconv_kernel(const float* __restrict__ wq, const float* __restrict__ wk,
                             const float* __restrict__ wv, const float* __restrict__ wg,
                             const float* __restrict__ wo, __bf16* __restrict__ dst) {
  int i = blockIdx.x * blockDim.x + threadIdx.x;
  if (i >= 5 * 160 * 32 / 5 * 5) return;  // 5120 threads
  int lane = i & 31;
  int t = i >> 5;                 // 0..159 = which*32 + kk*8 + nt
  if (t >= 160) return;
  int nt = t & 7, kk = (t >> 3) & 3, which = t >> 5;
  const float* w = (which == 0) ? wq : (which == 1) ? wk : (which == 2) ? wv
                 : (which == 3) ? wg : wo;
  int half = lane >> 4, nn = lane & 15;
#pragma unroll
  for (int e = 0; e < 16; ++e) {
    int k = kk * 32 + 16 * half + e;
    int n = nt * 16 + nn;
    dst[((long)t * 32 + lane) * 16 + e] = (__bf16)w[k * 128 + n];
  }
}

// ---------------- Kernel 1: LN + Q/K/V/G projections, swizzled outputs ----------------
__global__ __launch_bounds__(256) void proj_kernel(
    const float* __restrict__ tokens, const float* __restrict__ g1, const float* __restrict__ b1,
    const float* __restrict__ bg, const __bf16* __restrict__ Wsw,
    __bf16* __restrict__ QpA, float* __restrict__ Qf, __bf16* __restrict__ KpTA,
    __bf16* __restrict__ KpB, __bf16* __restrict__ VgB, float* __restrict__ ksum) {
  __shared__ __align__(32) float xf[64 * 128];
  __shared__ __align__(32) __bf16 xbA[16 * 512];  // [rt4][kp4] tiles, frag order
  __shared__ float ksumLds[128];
  int tid = threadIdx.x, lane = tid & 31, wave = tid >> 5, half = lane >> 4, nn = lane & 15;
  long t0 = (long)blockIdx.x * 64;
  long ch = blockIdx.x;  // one block == one 64-token chunk

  for (int i = tid; i < 64 * 128; i += 256) xf[i] = tokens[t0 * 128 + i];
  if (tid < 128) ksumLds[tid] = 0.f;
  __syncthreads();
  if (tid < 64) {
    float mu = 0.f;
    for (int c = 0; c < 128; ++c) mu += xf[tid * 128 + c];
    mu *= (1.f / 128.f);
    float m2 = 0.f;
    for (int c = 0; c < 128; ++c) { float d = xf[tid * 128 + c] - mu; m2 += d * d; }
    float rs = rsqrtf(m2 * (1.f / 128.f) + 1e-5f);
    int rt = tid >> 4, m = tid & 15;
    for (int c = 0; c < 128; ++c) {
      float v = (xf[tid * 128 + c] - mu) * rs * g1[c] + b1[c];
      int kp = c >> 5, k = c & 31;
      xbA[((rt * 4 + kp) * 32 + a_lane(m, k)) * 16 + a_elem(k)] = (__bf16)v;
    }
  }
  __syncthreads();

  for (int i = 0; i < 4; ++i) {
    int p = wave * 4 + i;
    int rt = p >> 3, ct = p & 7;
    v8f cq = {}, ck = {}, cv = {}, cg = {};
#pragma unroll
    for (int kk = 0; kk < 4; ++kk) {
      v16bf a = frag_load(&xbA[((rt * 4 + kk) * 32 + lane) * 16]);
      v16bf bq = frag_load(&Wsw[((long)(0 * 32 + kk * 8 + ct) * 32 + lane) * 16]);
      v16bf bk = frag_load(&Wsw[((long)(1 * 32 + kk * 8 + ct) * 32 + lane) * 16]);
      v16bf bv = frag_load(&Wsw[((long)(2 * 32 + kk * 8 + ct) * 32 + lane) * 16]);
      v16bf bgm = frag_load(&Wsw[((long)(3 * 32 + kk * 8 + ct) * 32 + lane) * 16]);
      cq = WMMA_BF16(a, bq, cq);
      ck = WMMA_BF16(a, bk, ck);
      cv = WMMA_BF16(a, bv, cv);
      cg = WMMA_BF16(a, bgm, cg);
    }
    int col = ct * 16 + nn;
    float bgc = bg[col];
    int kpc = col >> 5, kc = col & 31;           // QpA k-panel pieces
    int mt2 = col >> 4, m2 = col & 15;           // KpTA row-tile pieces (h dim)
    int kk3 = col >> 5, kl3 = col & 31;          // KpB k pieces (h dim)
    int nt4 = col >> 4;                          // VgB n-tile (h dim)
#pragma unroll
    for (int r = 0; r < 8; ++r) {
      int row = rt * 16 + r + 8 * half;          // token within chunk
      int m = r + 8 * half;
      float q = cq[r], k = ck[r];
      float qpv = q > 0.f ? q + 1.f : __expf(q);
      float kpv = k > 0.f ? k + 1.f : __expf(k);
      float gt = 1.f / (1.f + __expf(-(cg[r] + bgc)));
      float vgv = cv[r] * gt;
      long grt = ch * 4 + rt;  // global 16-row tile
      QpA[((grt * 4 + kpc) * 32 + a_lane(m, kc)) * 16 + a_elem(kc)] = (__bf16)qpv;
      Qf[(t0 + row) * 128 + col] = qpv;
      int kp2 = row >> 5, k2 = row & 31;         // KpTA: A elem (m2,h-tile) k=t
      KpTA[((ch * 16 + mt2 * 2 + kp2) * 32 + a_lane(m2, k2)) * 16 + a_elem(k2)] = (__bf16)kpv;
      int nt3 = row >> 4;                        // KpB: B elem k=h, n=t
      KpB[((ch * 16 + kk3 * 4 + nt3) * 32 + b_lane(row & 15, kl3)) * 16 + (kl3 & 15)] =
          (__bf16)kpv;
      int kp4 = row >> 5, kl4 = row & 31;        // VgB: B elem k=t, n=h
      VgB[((ch * 16 + kp4 * 8 + nt4) * 32 + b_lane(nn, kl4)) * 16 + (kl4 & 15)] = (__bf16)vgv;
      atomicAdd(&ksumLds[col], kpv);
    }
  }
  __syncthreads();
  if (tid < 128) ksum[ch * 128 + tid] = ksumLds[tid];
}

// ---------------- Kernel 2: per-chunk KV = Kp^T @ Vg (128x128 f32) ----------------
__global__ __launch_bounds__(256) void chunk_kv_kernel(
    const __bf16* __restrict__ KpTA, const __bf16* __restrict__ VgB, float* __restrict__ KV) {
  int tid = threadIdx.x, lane = tid & 31, wave = tid >> 5, half = lane >> 4, nn = lane & 15;
  long ch = blockIdx.x;
  for (int i = 0; i < 8; ++i) {
    int p = wave * 8 + i;
    int mt = p >> 3, nt = p & 7;
    v8f acc = {};
#pragma unroll
    for (int kk = 0; kk < 2; ++kk) {
      v16bf a = frag_load(&KpTA[((ch * 16 + mt * 2 + kk) * 32 + lane) * 16]);
      v16bf b = frag_load(&VgB[((ch * 16 + kk * 8 + nt) * 32 + lane) * 16]);
      acc = WMMA_BF16(a, b, acc);
    }
#pragma unroll
    for (int r = 0; r < 8; ++r) {
      int m = mt * 16 + r + 8 * half, n = nt * 16 + nn;
      KV[ch * 16384 + m * 128 + n] = acc[r];
    }
  }
}

// ---------------- Kernel 3: exclusive prefix scan; state stored B-frag-swizzled ----------------
__global__ __launch_bounds__(1024) void scan_kernel(
    const float* __restrict__ KV, const float* __restrict__ ksum,
    __bf16* __restrict__ SprevB, float* __restrict__ kprev) {
  int tid = threadIdx.x;
  long b = blockIdx.x;
  float s[16];
#pragma unroll
  for (int u = 0; u < 16; ++u) s[u] = 0.f;
  float ks = 0.f;
  for (int c = 0; c < 64; ++c) {
    long cb = b * 64 + c;
#pragma unroll
    for (int u = 0; u < 16; ++u) {
      int idx = tid * 16 + u;
      int k = idx >> 7, n = idx & 127;
      int kk = k >> 5, nt = n >> 4;
      long di = ((cb * 32 + kk * 8 + nt) * 32 + b_lane(n & 15, k & 31)) * 16 + (k & 15);
      SprevB[di] = (__bf16)s[u];  // exclusive prefix
    }
    if (tid < 128) kprev[cb * 128 + tid] = ks;
#pragma unroll
    for (int u = 0; u < 16; ++u) s[u] += KV[cb * 16384 + tid * 16 + u];
    if (tid < 128) ks += ksum[cb * 128 + tid];
  }
}

// ---------------- Kernel 4: attention output + out-proj + residual + LN ----------------
__global__ __launch_bounds__(256) void out_kernel(
    const float* __restrict__ tokens, const __bf16* __restrict__ QpA,
    const float* __restrict__ Qf, const __bf16* __restrict__ KpB,
    const __bf16* __restrict__ VgB, const __bf16* __restrict__ SprevB,
    const float* __restrict__ kprev, const __bf16* __restrict__ Wsw,
    const float* __restrict__ g2, const float* __restrict__ b2, float* __restrict__ out) {
  // LDS: [0,16K) P f32 -> later [0,32K) y f32 ; [16K,24K) P bf16 frag ;
  //      [32K,48K) attn bf16 frag ; [48K,+) denom + kprev
  __shared__ __align__(32) char smem[32768 + 16384 + 256 + 512];
  float* Pf = (float*)smem;
  __bf16* PbA = (__bf16*)(smem + 16384);     // [mt4][kp2] A-frag tiles
  __bf16* attnA = (__bf16*)(smem + 32768);   // [mt4][kp4] A-frag tiles
  float* den = (float*)(smem + 49152);
  float* kprevLds = (float*)(smem + 49152 + 256);
  float* yf = (float*)smem;

  int tid = threadIdx.x, lane = tid & 31, wave = tid >> 5, half = lane >> 4, nn = lane & 15;
  long t0 = (long)blockIdx.x * 64;
  long ch = blockIdx.x;

  if (tid < 128) kprevLds[tid] = kprev[ch * 128 + tid];

  // Stage 1: P = Qp @ Kp^T (64x64), causal mask, store f32 + A-frag bf16
  for (int i = 0; i < 2; ++i) {
    int p = wave * 2 + i;
    int mt = p >> 2, nt = p & 3;
    v8f acc = {};
#pragma unroll
    for (int kk = 0; kk < 4; ++kk) {
      v16bf a = frag_load(&QpA[(((ch * 4 + mt) * 4 + kk) * 32 + lane) * 16]);
      v16bf b = frag_load(&KpB[((ch * 16 + kk * 4 + nt) * 32 + lane) * 16]);
      acc = WMMA_BF16(a, b, acc);
    }
#pragma unroll
    for (int r = 0; r < 8; ++r) {
      int m = r + 8 * half;
      int mi = mt * 16 + m, nj = nt * 16 + nn;
      float v = (nj <= mi) ? acc[r] : 0.f;
      Pf[mi * 64 + nj] = v;
      int k = nj & 31;
      PbA[((mt * 2 + (nj >> 5)) * 32 + a_lane(m, k)) * 16 + a_elem(k)] = (__bf16)v;
    }
  }
  __syncthreads();

  // denominator: rowsum(P) + Qp . kprev (float4 loads)
  if (tid < 64) {
    float rs = 0.f;
    for (int j = 0; j < 64; ++j) rs += Pf[tid * 64 + j];
    const float4* q4 = (const float4*)&Qf[(t0 + tid) * 128];
    float dk = 0.f;
#pragma unroll 8
    for (int j = 0; j < 32; ++j) {
      float4 q = q4[j];
      dk += q.x * kprevLds[4 * j] + q.y * kprevLds[4 * j + 1] +
            q.z * kprevLds[4 * j + 2] + q.w * kprevLds[4 * j + 3];
    }
    den[tid] = fmaxf(rs + dk, 1e-6f);
  }
  __syncthreads();

  // Stage 2: numerator = P @ Vg + Qp @ Sprev ; divide ; -> attn A-frag
  for (int i = 0; i < 4; ++i) {
    int p = wave * 4 + i;
    int mt = p >> 3, nt = p & 7;
    __builtin_prefetch((const void*)&SprevB[((ch * 32 + nt) * 32 + lane) * 16], 0, 1);
    v8f acc = {};
#pragma unroll
    for (int kk = 0; kk < 2; ++kk) {  // intra, K = 64
      v16bf a = frag_load(&PbA[((mt * 2 + kk) * 32 + lane) * 16]);
      v16bf b = frag_load(&VgB[((ch * 16 + kk * 8 + nt) * 32 + lane) * 16]);
      acc = WMMA_BF16(a, b, acc);
    }
#pragma unroll
    for (int kk = 0; kk < 4; ++kk) {  // inter, K = 128
      v16bf a = frag_load(&QpA[(((ch * 4 + mt) * 4 + kk) * 32 + lane) * 16]);
      v16bf b = frag_load(&SprevB[((ch * 32 + kk * 8 + nt) * 32 + lane) * 16]);
      acc = WMMA_BF16(a, b, acc);
    }
#pragma unroll
    for (int r = 0; r < 8; ++r) {
      int m = r + 8 * half;
      int mi = mt * 16 + m, nj = nt * 16 + nn;
      int k = nj & 31;
      attnA[((mt * 4 + (nj >> 5)) * 32 + a_lane(m, k)) * 16 + a_elem(k)] =
          (__bf16)(acc[r] / den[mi]);
    }
  }
  __syncthreads();

  // Stage 3: y = tokens + 0.1 * (attn @ Wo)
  for (int i = 0; i < 4; ++i) {
    int p = wave * 4 + i;
    int mt = p >> 3, nt = p & 7;
    v8f acc = {};
#pragma unroll
    for (int kk = 0; kk < 4; ++kk) {
      v16bf a = frag_load(&attnA[((mt * 4 + kk) * 32 + lane) * 16]);
      v16bf b = frag_load(&Wsw[((long)(4 * 32 + kk * 8 + nt) * 32 + lane) * 16]);
      acc = WMMA_BF16(a, b, acc);
    }
#pragma unroll
    for (int r = 0; r < 8; ++r) {
      int mi = mt * 16 + r + 8 * half, nj = nt * 16 + nn;
      yf[mi * 128 + nj] = tokens[(t0 + mi) * 128 + nj] + 0.1f * acc[r];
    }
  }
  __syncthreads();

  // Stage 4: final layernorm
  if (tid < 64) {
    float mu = 0.f;
    for (int c = 0; c < 128; ++c) mu += yf[tid * 128 + c];
    mu *= (1.f / 128.f);
    float m2 = 0.f;
    for (int c = 0; c < 128; ++c) { float d = yf[tid * 128 + c] - mu; m2 += d * d; }
    float rsv = rsqrtf(m2 * (1.f / 128.f) + 1e-5f);
    for (int c = 0; c < 128; ++c)
      out[(t0 + tid) * 128 + c] = (yf[tid * 128 + c] - mu) * rsv * g2[c] + b2[c];
  }
}

extern "C" void kernel_launch(void* const* d_in, const int* in_sizes, int n_in,
                              void* d_out, int out_size, void* d_ws, size_t ws_size,
                              hipStream_t stream) {
  (void)in_sizes; (void)n_in; (void)out_size; (void)ws_size;
  const float* tokens = (const float*)d_in[0];
  const float* Wq = (const float*)d_in[1];
  const float* Wk = (const float*)d_in[2];
  const float* Wv = (const float*)d_in[3];
  const float* Wg = (const float*)d_in[4];
  const float* bg = (const float*)d_in[5];
  const float* Wo = (const float*)d_in[6];
  const float* g1 = (const float*)d_in[7];
  const float* b1 = (const float*)d_in[8];
  const float* g2 = (const float*)d_in[9];
  const float* b2 = (const float*)d_in[10];
  float* out = (float*)d_out;

  char* ws = (char*)d_ws;
  size_t off = 0;
  auto alloc = [&](size_t bytes) { size_t p = off; off += (bytes + 255) & ~(size_t)255; return p; };
  const long BT = 2L * 4096;   // 8192 token rows
  const long NCH = 128;        // B * T/64 chunks

  __bf16* Wsw   = (__bf16*)(ws + alloc(160 * 512 * 2));          // 5 weights, B-frag
  __bf16* QpA   = (__bf16*)(ws + alloc((BT / 16) * 4 * 512 * 2)); // A-frag, 4 k-panels
  float*  Qf    = (float*)(ws + alloc(BT * 128 * 4));             // row-major f32
  __bf16* KpTA  = (__bf16*)(ws + alloc(NCH * 16 * 512 * 2));      // Kp^T A-frag
  __bf16* KpB   = (__bf16*)(ws + alloc(NCH * 16 * 512 * 2));      // Kp^T B-frag (k=h,n=t)
  __bf16* VgB   = (__bf16*)(ws + alloc(NCH * 16 * 512 * 2));      // Vg B-frag
  float*  KV    = (float*)(ws + alloc(NCH * 16384 * 4));
  float*  ksum  = (float*)(ws + alloc(NCH * 128 * 4));
  __bf16* SprevB= (__bf16*)(ws + alloc(NCH * 32 * 512 * 2));      // state B-frag
  float*  kprev = (float*)(ws + alloc(NCH * 128 * 4));

  wconv_kernel<<<20, 256, 0, stream>>>(Wq, Wk, Wv, Wg, Wo, Wsw);
  proj_kernel<<<128, 256, 0, stream>>>(tokens, g1, b1, bg, Wsw, QpA, Qf, KpTA, KpB, VgB, ksum);
  chunk_kv_kernel<<<128, 256, 0, stream>>>(KpTA, VgB, KV);
  scan_kernel<<<2, 1024, 0, stream>>>(KV, ksum, SprevB, kprev);
  out_kernel<<<128, 256, 0, stream>>>(tokens, QpA, Qf, KpB, VgB, SprevB, kprev, Wsw, g2, b2, out);
}
